// SwinTransformerBlock_86749749445094
// MI455X (gfx1250) — matrix operand order repaired and checked by reading, not compiled, not run on hardware
//
#include <hip/hip_runtime.h>
#include <hip/hip_fp16.h>
#include <stdint.h>
#include <stddef.h>

// ---------------------------------------------------------------------------
// Swin Transformer block for MI455X (gfx1250):
//   - all matmuls through v_wmma_f32_16x16x32_f16 (f16 in, f32 accumulate)
//   - GEMM tile staging via GLOBAL_LOAD_ASYNC_TO_LDS_B128 when available
// ---------------------------------------------------------------------------

typedef _Float16 h16;
typedef __attribute__((ext_vector_type(16))) _Float16 v16h;
typedef __attribute__((ext_vector_type(8)))  float    v8f;
typedef int v4i __attribute__((vector_size(16)));   // matches builtin param type

#define AS1 __attribute__((address_space(1)))
#define AS3 __attribute__((address_space(3)))

#if defined(__AMDGCN__) && __has_builtin(__builtin_amdgcn_global_load_async_to_lds_b128)
#define HAVE_ASYNC_LDS 1
#else
#define HAVE_ASYNC_LDS 0
#endif

__device__ inline void wait_async0() {
#if __has_builtin(__builtin_amdgcn_s_wait_asynccnt)
  __builtin_amdgcn_s_wait_asynccnt(0);
#else
  asm volatile("s_wait_asynccnt 0" ::: "memory");
#endif
}

#define NB     16
#define NHW    56
#define NC     384
#define NHEAD  12
#define NWSZ   7
#define NTOK   49          // tokens per window
#define NHD    32          // head dim
#define NHID   1536
#define NWIN_PER_IMG 64
#define TOT_WIN (NB * NWIN_PER_IMG)          // 1024
#define UNITS   (TOT_WIN * NHEAD)            // 12288
#define MTOK    (TOT_WIN * NTOK)             // 50176
#define QKSCALE 0.17677669529663689f         // 32^-0.5

// ---------------------------------------------------------------------------
// WMMA fragment loaders (wave32).  contiguous K order, A and B alike:
//   A 16x32 f16:  lane L -> row M=L&15;  K base = (L>=16)?16:0;
//                 vgpr i (0..7) : K = base+2i, base+2i+1
//   B 32x16 f16 (same K order as A): lane L -> col N=L&15;
//                 lanes 0-15 hold K=0..15, lanes 16-31 hold K=16..31.
// Both read contiguous K pairs -> dword loads from a row-major [row][K] array.
// ---------------------------------------------------------------------------
__device__ inline v16h load_frag_a(const h16* __restrict__ base, int row0,
                                   int stride, int rowClamp) {
  const int lane = threadIdx.x & 31;
  int row = row0 + (lane & 15);
  row = (row > rowClamp) ? rowClamp : row;
  const int kb = (lane >> 4) << 4;          // lanes 16-31 hold K=16..31
  const h16* p = base + (size_t)row * stride;
  union { v16h v; uint32_t u[8]; } f;
#pragma unroll
  for (int i = 0; i < 8; ++i) {
    const int k = kb + 2 * i;
    f.u[i] = *(const uint32_t*)(p + k);
  }
  return f.v;
}

__device__ inline v16h load_frag_b(const h16* __restrict__ base, int row0,
                                   int stride, int rowClamp) {
  const int lane = threadIdx.x & 31;
  int row = row0 + (lane & 15);             // "row" of array == N column of B
  row = (row > rowClamp) ? rowClamp : row;
  const int kb = (lane >> 4) << 4;          // lanes 16-31 hold K=16..31
  const h16* p = base + (size_t)row * stride;
  union { v16h v; uint32_t u[8]; } f;
#pragma unroll
  for (int i = 0; i < 8; ++i)
    f.u[i] = *(const uint32_t*)(p + kb + 2 * i);
  return f.v;
}

__device__ inline v8f wmma_f16(v16h a, v16h b, v8f c) {
  return __builtin_amdgcn_wmma_f32_16x16x32_f16(false, a, false, b,
                                                (short)0, c, false, false);
}

// ---------------------------------------------------------------------------
// Prep kernels
// ---------------------------------------------------------------------------
__global__ void k_f32_to_f16(const float* __restrict__ src, h16* __restrict__ dst, int n) {
  for (int i = blockIdx.x * blockDim.x + threadIdx.x; i < n; i += gridDim.x * blockDim.x)
    dst[i] = (h16)src[i];
}

__global__ void k_zero_f16(h16* __restrict__ dst, int n) {
  for (int i = blockIdx.x * blockDim.x + threadIdx.x; i < n; i += gridDim.x * blockDim.x)
    dst[i] = (h16)0.0f;
}

// biasT[h][n][m] = rpb_table[RPI(n,m)][h]
__global__ void k_build_bias(const float* __restrict__ rpb, float* __restrict__ biasT) {
  int e = blockIdx.x * blockDim.x + threadIdx.x;
  if (e >= NHEAD * NTOK * NTOK) return;
  int h = e / (NTOK * NTOK);
  int rem = e % (NTOK * NTOK);
  int n = rem / NTOK, m = rem % NTOK;
  int in_ = n / NWSZ, jn = n % NWSZ;
  int im  = m / NWSZ, jm = m % NWSZ;
  int idx = (in_ - im + NWSZ - 1) * (2 * NWSZ - 1) + (jn - jm + NWSZ - 1);
  biasT[e] = rpb[idx * NHEAD + h];
}

// shift (-3,-3) + window partition -> xw f16 [MTOK][NC]
__global__ void k_shift_partition(const float* __restrict__ x, h16* __restrict__ xw) {
  const int total = MTOK * NC;
  for (int e = blockIdx.x * blockDim.x + threadIdx.x; e < total; e += gridDim.x * blockDim.x) {
    int c = e % NC;
    int t = e / NC;
    int n = t % NTOK;
    int win = t / NTOK;
    int b  = win >> 6;
    int wi = win & 63;
    int wh = wi >> 3, ww = wi & 7;
    int r = n / NWSZ, s = n % NWSZ;
    int hh = wh * NWSZ + r + 3; if (hh >= NHW) hh -= NHW;   // roll(-3) source
    int w2 = ww * NWSZ + s + 3; if (w2 >= NHW) w2 -= NHW;
    xw[e] = (h16)x[(((size_t)b * NHW + hh) * NHW + w2) * NC + c];
  }
}

// ---------------------------------------------------------------------------
// Generic WMMA GEMM:  C[M,N] = A[M,K] * W[N,K]^T  (+ mode-specific epilogue)
//   mode 0: QKV   -> split into scaled Q, K, transposed/padded V (f16)
//   mode 1: proj  -> window-reverse + unshift scatter into x2 (f32)
//   mode 2: fc1   -> bias + exact GELU -> f16
//   mode 3: fc2   -> bias + residual   -> f32 (d_out)
// block = 256 threads (8 waves), tile 128x64, K-step 32, wave tile 32x32.
// ---------------------------------------------------------------------------
#define BM 128
#define BN 64
#define BK 32

__global__ __launch_bounds__(256) void gemm_wmma_kernel(
    const h16* __restrict__ A, const h16* __restrict__ Bw,
    int M, int N, int K, int mode,
    const float* __restrict__ bvec,
    h16* __restrict__ oq, h16* __restrict__ ok, h16* __restrict__ ovT,
    float* __restrict__ of32, h16* __restrict__ of16,
    const float* __restrict__ resid, float* __restrict__ ofinal) {
  __shared__ __align__(16) h16 As[BM * BK];   // 8 KB
  __shared__ __align__(16) h16 Bs[BN * BK];   // 4 KB

  const int tid   = threadIdx.x;
  const int lane  = tid & 31;
  const int wave  = tid >> 5;
  const int waveM = wave & 3;   // 4 waves along M
  const int waveN = wave >> 2;  // 2 waves along N
  const int gm0   = blockIdx.x * BM;
  const int gn0   = blockIdx.y * BN;

  v8f acc[2][2];
#pragma unroll
  for (int i = 0; i < 2; ++i)
#pragma unroll
    for (int j = 0; j < 2; ++j) acc[i][j] = {};

  for (int k0 = 0; k0 < K; k0 += BK) {
#if HAVE_ASYNC_LDS
    // Async DMA: global -> LDS without VGPR round trip (ASYNCcnt tracked)
#pragma unroll
    for (int it = 0; it < 2; ++it) {
      int i = tid + it * 256;
      int row = i >> 2, seg = i & 3;
      __builtin_amdgcn_global_load_async_to_lds_b128(
          (AS1 v4i*)(A + (size_t)(gm0 + row) * K + k0 + seg * 8),
          (AS3 v4i*)(As + i * 8), 0, 0);
    }
    {
      int row = tid >> 2, seg = tid & 3;
      __builtin_amdgcn_global_load_async_to_lds_b128(
          (AS1 v4i*)(Bw + (size_t)(gn0 + row) * K + k0 + seg * 8),
          (AS3 v4i*)(Bs + tid * 8), 0, 0);
    }
    wait_async0();
#else
    // Fallback: stage via VGPRs (uint4 chunks of 8 halves)
#pragma unroll
    for (int it = 0; it < 2; ++it) {
      int i = tid + it * 256;
      int row = i >> 2, seg = i & 3;
      ((uint4*)As)[i] = *(const uint4*)(A + (size_t)(gm0 + row) * K + k0 + seg * 8);
    }
    {
      int row = tid >> 2, seg = tid & 3;
      ((uint4*)Bs)[tid] = *(const uint4*)(Bw + (size_t)(gn0 + row) * K + k0 + seg * 8);
    }
#endif
    __syncthreads();

    v16h af[2], bf[2];
#pragma unroll
    for (int i = 0; i < 2; ++i)
      af[i] = load_frag_a(As, waveM * 32 + i * 16, BK, BM - 1);
#pragma unroll
    for (int j = 0; j < 2; ++j)
      bf[j] = load_frag_b(Bs, waveN * 32 + j * 16, BK, BN - 1);
#pragma unroll
    for (int i = 0; i < 2; ++i)
#pragma unroll
      for (int j = 0; j < 2; ++j)
        acc[i][j] = wmma_f16(af[i], bf[j], acc[i][j]);
    __syncthreads();
  }

  // ---------------- epilogue ----------------
#pragma unroll
  for (int i = 0; i < 2; ++i) {
#pragma unroll
    for (int j = 0; j < 2; ++j) {
      const int rowBase = gm0 + waveM * 32 + i * 16;
      const int colBase = gn0 + waveN * 32 + j * 16;
      const int col = colBase + (lane & 15);
      const float bv = bvec[col];
#pragma unroll
      for (int r = 0; r < 8; ++r) {
        const int row = rowBase + r + ((lane >> 4) << 3);
        float val = acc[i][j][r] + bv;
        if (mode == 0) {
          int which = col / NC;
          int hh = (col % NC) / NHD;
          int d  = col % NHD;
          int win = row / NTOK, n = row % NTOK;
          size_t unit = (size_t)win * NHEAD + hh;
          if (which == 0)      oq[(unit * NTOK + n) * NHD + d] = (h16)(val * QKSCALE);
          else if (which == 1) ok[(unit * NTOK + n) * NHD + d] = (h16)val;
          else                 ovT[unit * (NHD * 64) + d * 64 + n] = (h16)val;
        } else if (mode == 1) {
          int win = row / NTOK, n = row % NTOK;
          int b  = win >> 6, wi = win & 63;
          int wh = wi >> 3, ww = wi & 7;
          int rr = n / NWSZ, ss = n % NWSZ;
          int h2 = wh * NWSZ + rr + 3; if (h2 >= NHW) h2 -= NHW;   // roll(+3)
          int w2 = ww * NWSZ + ss + 3; if (w2 >= NHW) w2 -= NHW;
          of32[(((size_t)b * NHW + h2) * NHW + w2) * NC + col] = val;
        } else if (mode == 2) {
          float ge = 0.5f * val * (1.0f + erff(val * 0.70710678118654752f));
          of16[(size_t)row * N + col] = (h16)ge;
        } else {
          size_t idx = (size_t)row * NC + col;
          ofinal[idx] = resid[idx] + val;
        }
      }
    }
  }
}

// ---------------------------------------------------------------------------
// Attention: one wave per (window, head); 2 waves per block.
//   S = Q K^T (16 WMMAs) -> +bias +mask -> softmax -> P f16 -> O = P V (16 WMMAs)
// ---------------------------------------------------------------------------
__global__ __launch_bounds__(64) void attn_kernel(
    const h16* __restrict__ q, const h16* __restrict__ k, const h16* __restrict__ vT,
    const float* __restrict__ biasT, const float* __restrict__ mask,
    h16* __restrict__ attout) {
  __shared__ float S[2][64 * 64];   // 32 KB
  __shared__ h16   P[2][64 * 64];   // 16 KB

  const int wid  = threadIdx.x >> 5;
  const int lane = threadIdx.x & 31;
  const int unit = blockIdx.x * 2 + wid;
  const int win  = unit / NHEAD;
  const int head = unit % NHEAD;
  const int wi   = win & 63;

  const h16* qb = q  + (size_t)unit * NTOK * NHD;
  const h16* kb = k  + (size_t)unit * NTOK * NHD;
  const h16* vb = vT + (size_t)unit * NHD * 64;
  float* Sw = S[wid];
  h16*   Pw = P[wid];

  // ---- S = Q K^T  (M=N padded to 64, K=32 single step) ----
  v16h aq[4];
#pragma unroll
  for (int mt = 0; mt < 4; ++mt)
    aq[mt] = load_frag_a(qb, mt * 16, NHD, NTOK - 1);
#pragma unroll
  for (int nt = 0; nt < 4; ++nt) {
    v16h bk = load_frag_b(kb, nt * 16, NHD, NTOK - 1);
#pragma unroll
    for (int mt = 0; mt < 4; ++mt) {
      v8f acc = {};
      acc = wmma_f16(aq[mt], bk, acc);
#pragma unroll
      for (int r = 0; r < 8; ++r)
        Sw[(mt * 16 + r + ((lane >> 4) << 3)) * 64 + nt * 16 + (lane & 15)] = acc[r];
    }
  }
  __syncthreads();

  // ---- softmax rows (rows 0..48) with rel-pos bias + shift mask ----
  const float* bh = biasT + (size_t)head * NTOK * NTOK;
  const float* mw = mask  + (size_t)wi   * NTOK * NTOK;
  for (int r = lane; r < NTOK; r += 32) {
    float mx = -1e30f;
    for (int c = 0; c < NTOK; ++c) {
      float v = Sw[r * 64 + c] + bh[r * NTOK + c] + mw[r * NTOK + c];
      Sw[r * 64 + c] = v;
      mx = fmaxf(mx, v);
    }
    float sum = 0.0f;
    for (int c = 0; c < NTOK; ++c) {
      float e = __expf(Sw[r * 64 + c] - mx);
      Sw[r * 64 + c] = e;
      sum += e;
    }
    float inv = 1.0f / sum;
    for (int c = 0; c < NTOK; ++c) Pw[r * 64 + c] = (h16)(Sw[r * 64 + c] * inv);
    for (int c = NTOK; c < 64; ++c) Pw[r * 64 + c] = (h16)0.0f;   // pad K dim
  }
  __syncthreads();

  // ---- O = P V  (M=64, K=64 in two steps, N=32) ----
#pragma unroll
  for (int nt = 0; nt < 2; ++nt) {
#pragma unroll
    for (int mt = 0; mt < 4; ++mt) {
      v8f acc = {};
#pragma unroll
      for (int kt = 0; kt < 2; ++kt) {
        v16h a = load_frag_a(Pw + kt * 32, mt * 16, 64, 63);
        v16h b = load_frag_b(vb + kt * 32, nt * 16, 64, NHD - 1);
        acc = wmma_f16(a, b, acc);
      }
      const int cold = head * NHD + nt * 16 + (lane & 15);
#pragma unroll
      for (int r = 0; r < 8; ++r) {
        int row = mt * 16 + r + ((lane >> 4) << 3);
        if (row < NTOK)
          attout[((size_t)win * NTOK + row) * NC + cold] = (h16)acc[r];
      }
    }
  }
}

// ---------------------------------------------------------------------------
// LayerNorm over C=384, one wave32 per token, output f16 for fc1
// ---------------------------------------------------------------------------
__global__ __launch_bounds__(256) void layernorm_kernel(
    const float* __restrict__ x2, const float* __restrict__ g,
    const float* __restrict__ b, h16* __restrict__ xn) {
  const int lane = threadIdx.x & 31;
  const int wid  = threadIdx.x >> 5;
  const size_t tok = (size_t)blockIdx.x * 8 + wid;
  const float* xp = x2 + tok * NC;
  float s = 0.0f, ss = 0.0f;
  float vals[12];
#pragma unroll
  for (int i = 0; i < 12; ++i) {
    float v = xp[lane + i * 32];
    vals[i] = v; s += v; ss += v * v;
  }
#pragma unroll
  for (int off = 16; off > 0; off >>= 1) {
    s  += __shfl_xor(s,  off);
    ss += __shfl_xor(ss, off);
  }
  const float mu  = s * (1.0f / NC);
  const float var = ss * (1.0f / NC) - mu * mu;
  const float inv = rsqrtf(var + 1e-5f);
  h16* op = xn + tok * NC;
#pragma unroll
  for (int i = 0; i < 12; ++i) {
    int c = lane + i * 32;
    op[c] = (h16)((vals[i] - mu) * inv * g[c] + b[c]);
  }
}

// ---------------------------------------------------------------------------
// Host side
// ---------------------------------------------------------------------------
extern "C" void kernel_launch(void* const* d_in, const int* in_sizes, int n_in,
                              void* d_out, int out_size, void* d_ws, size_t ws_size,
                              hipStream_t stream) {
  (void)in_sizes; (void)n_in; (void)out_size; (void)ws_size;
  const float* x       = (const float*)d_in[0];
  const float* maskm   = (const float*)d_in[1];
  const float* qkv_w   = (const float*)d_in[2];
  const float* qkv_b   = (const float*)d_in[3];
  const float* proj_w  = (const float*)d_in[4];
  const float* proj_b  = (const float*)d_in[5];
  const float* rpb     = (const float*)d_in[6];
  const float* norm2_g = (const float*)d_in[7];
  const float* norm2_b = (const float*)d_in[8];
  const float* fc1_w   = (const float*)d_in[9];
  const float* fc1_b   = (const float*)d_in[10];
  const float* fc2_w   = (const float*)d_in[11];
  const float* fc2_b   = (const float*)d_in[12];
  float* out = (float*)d_out;

  // workspace carve-up (256B aligned)
  char* w = (char*)d_ws;
  size_t off = 0;
  auto carve = [&](size_t bytes) -> void* {
    void* p = w + off;
    off = (off + bytes + 255) & ~(size_t)255;
    return p;
  };
  const size_t HTOK = (size_t)MTOK * NC;                 // 50176*384
  h16*  xw_attout = (h16*)carve(HTOK * sizeof(h16));     // xw, later reused as attn out
  h16*  q_xn      = (h16*)carve(HTOK * sizeof(h16));     // Q, later reused as norm out
  h16*  kbuf      = (h16*)carve(HTOK * sizeof(h16));
  h16*  vT        = (h16*)carve((size_t)UNITS * NHD * 64 * sizeof(h16));
  h16*  wq16      = (h16*)carve((size_t)3 * NC * NC * sizeof(h16));
  h16*  wp16      = (h16*)carve((size_t)NC * NC * sizeof(h16));
  h16*  w116      = (h16*)carve((size_t)NHID * NC * sizeof(h16));
  h16*  w216      = (h16*)carve((size_t)NC * NHID * sizeof(h16));
  float* biasT    = (float*)carve((size_t)NHEAD * NTOK * NTOK * sizeof(float));
  float* x2       = (float*)carve(HTOK * sizeof(float));
  h16*  h1        = (h16*)carve((size_t)MTOK * NHID * sizeof(h16));

  // ---- prep ----
  k_f32_to_f16<<<1024, 256, 0, stream>>>(qkv_w,  wq16, 3 * NC * NC);
  k_f32_to_f16<<<512,  256, 0, stream>>>(proj_w, wp16, NC * NC);
  k_f32_to_f16<<<1024, 256, 0, stream>>>(fc1_w,  w116, NHID * NC);
  k_f32_to_f16<<<1024, 256, 0, stream>>>(fc2_w,  w216, NC * NHID);
  k_zero_f16  <<<2048, 256, 0, stream>>>(vT, UNITS * NHD * 64);
  k_build_bias<<<(NHEAD * NTOK * NTOK + 255) / 256, 256, 0, stream>>>(rpb, biasT);
  k_shift_partition<<<4096, 256, 0, stream>>>(x, xw_attout);

  // ---- QKV GEMM: [50176,384] x [1152,384]^T ----
  gemm_wmma_kernel<<<dim3(MTOK / BM, (3 * NC) / BN), 256, 0, stream>>>(
      xw_attout, wq16, MTOK, 3 * NC, NC, 0, qkv_b,
      q_xn, kbuf, vT, nullptr, nullptr, nullptr, nullptr);

  // ---- windowed attention ----
  attn_kernel<<<UNITS / 2, 64, 0, stream>>>(q_xn, kbuf, vT, biasT, maskm, xw_attout);

  // ---- proj GEMM + window-reverse + unshift -> x2 ----
  gemm_wmma_kernel<<<dim3(MTOK / BM, NC / BN), 256, 0, stream>>>(
      xw_attout, wp16, MTOK, NC, NC, 1, proj_b,
      nullptr, nullptr, nullptr, x2, nullptr, nullptr, nullptr);

  // ---- LayerNorm -> f16 ----
  layernorm_kernel<<<MTOK / 8, 256, 0, stream>>>(x2, norm2_g, norm2_b, q_xn);

  // ---- fc1 GEMM + GELU ----
  gemm_wmma_kernel<<<dim3(MTOK / BM, NHID / BN), 256, 0, stream>>>(
      q_xn, w116, MTOK, NHID, NC, 2, fc1_b,
      nullptr, nullptr, nullptr, nullptr, h1, nullptr, nullptr);

  // ---- fc2 GEMM + residual -> d_out ----
  gemm_wmma_kernel<<<dim3(MTOK / BM, NC / BN), 256, 0, stream>>>(
      h1, w216, MTOK, NC, NHID, 3, fc2_b,
      nullptr, nullptr, nullptr, nullptr, nullptr, x2, out);
}
